// TransformerLM_85710367359168
// MI455X (gfx1250) — matrix operand order repaired and checked by reading, not compiled, not run on hardware
//
#include <hip/hip_runtime.h>
#include <hip/hip_bf16.h>
#include <stdint.h>

typedef _Float16 v16h __attribute__((ext_vector_type(16)));
typedef _Float16 v8h  __attribute__((ext_vector_type(8)));
typedef float    v8f  __attribute__((ext_vector_type(8)));
typedef float    v4f  __attribute__((ext_vector_type(4)));

constexpr int cB = 2, cS = 1024, cL = 6, cD = 1024, cH = 16, cDK = 64, cFF = 4096, cV = 32000;
constexpr int BM = 128, BN = 128, BK = 32, PAD = 8;  // LDS row stride 40 halves = 80B
constexpr int NT = 256;

// Build a 16-half fragment from two 16-byte LDS chunks (ISA 7.12.2 layouts).
__device__ __forceinline__ v16h frag_from(const _Float16* p0, const _Float16* p1) {
  v8h x0 = *(const v8h*)p0;
  v8h x1 = *(const v8h*)p1;
  return __builtin_shufflevector(x0, x1, 0, 1, 2, 3, 4, 5, 6, 7, 8, 9, 10, 11, 12, 13, 14, 15);
}

// Async global->LDS 16B copy per lane (ASYNCcnt-tracked; ISA 10.4 / 15.18.3).
// LDS byte offset = low 32 bits of the generic shared-pointer address.
__device__ __forceinline__ void async_copy_b128(const _Float16* lds_dst,
                                                const _Float16* gsrc) {
  unsigned lds_off = (unsigned)(uintptr_t)lds_dst;
  asm volatile("global_load_async_to_lds_b128 %0, %1, off"
               :
               : "v"(lds_off), "v"(gsrc)
               : "memory");
}
__device__ __forceinline__ void wait_async() {
  asm volatile("s_wait_asynccnt 0x0" : : : "memory");
}

// ---------------------------------------------------------------------------
// WMMA GEMM: C[M,N] = epi(A16[M,K] @ Bf32[K,N] + bias [+ Res_f32])
//   A (f16) staged via async global->LDS copies; B (f32 weights) converted to
//   f16 on the LDS store (each weight used once per pass).
//   128x128 block tile, 8 waves (2x4), 64x32 per wave = 8 WMMA / wave / k-step.
//   LDS double-buffered: stage tile s+1 while computing tile s; one barrier,
//   preceded by s_wait_asynccnt 0 so async A-tile writes are LDS-visible.
//   EPI: 0 = +bias, 1 = +bias,ReLU, 2 = +bias+Res.  F16OUT: C stored as f16.
// ---------------------------------------------------------------------------
template <int EPI, bool F16OUT>
__global__ __launch_bounds__(NT) void gemm_wmma_k(
    const _Float16* __restrict__ A, int lda, const float* __restrict__ Bw, int ldb,
    void* __restrict__ Cv, int ldc, const float* __restrict__ bias,
    const float* __restrict__ Res, int K) {
  __shared__ __align__(16) _Float16 As[2][BM][BK + PAD];
  __shared__ __align__(16) _Float16 Bs[2][BN][BK + PAD];

  const int m0 = blockIdx.x * BM;
  const int n0 = blockIdx.y * BN;
  const int tid = threadIdx.x;
  const int lane = tid & 31, wid = tid >> 5;
  const int wm = wid >> 2, wn = wid & 3;  // 2x4 wave grid, 64x32 per wave
  const int mr = lane & 15, kh = lane >> 4;

  auto stage = [&](int k0, int buf) {
    // A tile 128x32 f16: async global->LDS b128 copies (no VGPR round-trip).
#pragma unroll
    for (int i = 0; i < 2; i++) {
      int idx = tid + i * NT;  // 512 x 16B
      int r = idx >> 2, c = (idx & 3) << 3;
      async_copy_b128(&As[buf][r][c], A + (size_t)(m0 + r) * lda + k0 + c);
    }
    // B tile 32x128 f32 -> f16, transposed: Bs[n][k].
#pragma unroll
    for (int i = 0; i < 4; i++) {
      int idx = tid + i * NT;                 // 1024 float4
      int r = idx >> 5, c = (idx & 31) << 2;  // r = k row, c = n col base
      v4f val = *(const v4f*)(Bw + (size_t)(k0 + r) * ldb + n0 + c);
      Bs[buf][c + 0][r] = (_Float16)val[0];
      Bs[buf][c + 1][r] = (_Float16)val[1];
      Bs[buf][c + 2][r] = (_Float16)val[2];
      Bs[buf][c + 3][r] = (_Float16)val[3];
    }
  };

  v8f acc[4][2] = {};

  stage(0, 0);
  wait_async();
  __syncthreads();
  const int steps = K / BK;
  for (int s = 0; s < steps; s++) {
    const int buf = s & 1;
    if (s + 1 < steps) stage((s + 1) * BK, buf ^ 1);
    if (s + 2 < steps)  // L2 prefetch of the tile after next (global_prefetch_b8)
      __builtin_prefetch(Bw + (size_t)((s + 2) * BK + (tid & 31)) * ldb + n0, 0, 3);

    v16h af[4], bf[2];
#pragma unroll
    for (int i = 0; i < 4; i++) {
      const _Float16* ar = &As[buf][wm * 64 + i * 16 + mr][0];
      af[i] = frag_from(ar + kh * 8, ar + 16 + kh * 8);
    }
#pragma unroll
    for (int j = 0; j < 2; j++) {
      const _Float16* br = &Bs[buf][wn * 32 + j * 16 + mr][0];
      bf[j] = frag_from(br + kh * 8, br + 16 + kh * 8);
    }
#pragma unroll
    for (int i = 0; i < 4; i++)
#pragma unroll
      for (int j = 0; j < 2; j++)
        acc[i][j] = __builtin_amdgcn_wmma_f32_16x16x32_f16(
            false, af[i], false, bf[j], (short)0, acc[i][j], false, false);
    wait_async();
    __syncthreads();
  }

#pragma unroll
  for (int i = 0; i < 4; i++) {
#pragma unroll
    for (int j = 0; j < 2; j++) {
      const int gmb = m0 + wm * 64 + i * 16 + kh * 8;
      const int gn = n0 + wn * 32 + j * 16 + mr;
      const float bval = bias ? bias[gn] : 0.f;
#pragma unroll
      for (int r = 0; r < 8; r++) {
        float v = acc[i][j][r] + bval;
        if constexpr (EPI == 1) v = fmaxf(v, 0.f);
        if constexpr (EPI == 2) v += Res[(size_t)(gmb + r) * ldc + gn];
        if constexpr (F16OUT)
          ((_Float16*)Cv)[(size_t)(gmb + r) * ldc + gn] = (_Float16)v;
        else
          ((float*)Cv)[(size_t)(gmb + r) * ldc + gn] = v;
      }
    }
  }
}

// ---------------------------------------------------------------------------
// Fused causal flash attention, all-f16 operands (f32 softmax/accumulators).
//   Grid: (S/128, B*H). 8 waves; each wave owns 16 q-rows and the full 64-wide
//   k/dk extent so online-softmax stats stay wave-local. Q/K tiles staged via
//   async global->LDS copies; V transposed through VGPRs.
// ---------------------------------------------------------------------------
constexpr int FA_BQ = 128;
constexpr int FA_BK = 64;
constexpr int FPAD = 8;

__global__ __launch_bounds__(NT) void flash_attn_k(const _Float16* __restrict__ Q,
                                                   const _Float16* __restrict__ Km,
                                                   const _Float16* __restrict__ Vm,
                                                   _Float16* __restrict__ O) {
  __shared__ __align__(16) _Float16 QP[FA_BQ][cDK + FPAD];  // Q tile, reused for P
  __shared__ __align__(16) _Float16 Kt[FA_BK][cDK + FPAD];  // K rows  [n][k]
  __shared__ __align__(16) _Float16 Vt[cDK][FA_BK + FPAD];  // V^T     [dk][k]

  const int z = blockIdx.y;
  const int b = z / cH, h = z % cH;
  const int q0 = blockIdx.x * FA_BQ;
  const int tid = threadIdx.x, lane = tid & 31, wr = tid >> 5;
  const int mr = lane & 15, hi = lane >> 4;

  const _Float16* Qp = Q + (size_t)b * cS * cD + h * cDK;
  const _Float16* Kp = Km + (size_t)b * cS * cD + h * cDK;
  const _Float16* Vp = Vm + (size_t)b * cS * cD + h * cDK;

  // Q tile 128x64 f16: async copies.
#pragma unroll
  for (int i = 0; i < 4; i++) {
    int idx = tid + i * NT;  // 1024 x 16B
    int r = idx >> 3, c = (idx & 7) << 3;
    async_copy_b128(&QP[r][c], Qp + (size_t)(q0 + r) * cD + c);
  }
  wait_async();
  __syncthreads();

  v16h aq[2];
  {
    const _Float16* qr = &QP[wr * 16 + mr][0];
    aq[0] = frag_from(qr + hi * 8, qr + 16 + hi * 8);
    aq[1] = frag_from(qr + 32 + hi * 8, qr + 48 + hi * 8);
  }
  __syncthreads();  // QP may now be reused for P

  v8f po[4] = {};
  float mrow[8], lrow[8];
#pragma unroll
  for (int r = 0; r < 8; r++) {
    mrow[r] = -1.0e30f;
    lrow[r] = 0.f;
  }

  const int ktiles = (q0 + FA_BQ) / FA_BK;
  for (int kt = 0; kt < ktiles; kt++) {
    const int kc0 = kt * FA_BK;
    __syncthreads();
    // K tile 64x64 f16 -> Kt[n][k]: async copies.
#pragma unroll
    for (int i = 0; i < 2; i++) {
      int idx = tid + i * NT;  // 512 x 16B
      int r = idx >> 3, c = (idx & 7) << 3;
      async_copy_b128(&Kt[r][c], Kp + (size_t)(kc0 + r) * cD + c);
    }
    // V tile 64(k) x 64(dk) f16 -> Vt[dk][k] (transpose through VGPRs).
#pragma unroll
    for (int i = 0; i < 2; i++) {
      int idx = tid + i * NT;
      int r = idx >> 3, c = (idx & 7) << 3;
      v8h hv = *(const v8h*)(Vp + (size_t)(kc0 + r) * cD + c);
#pragma unroll
      for (int jj = 0; jj < 8; jj++) Vt[c + jj][r] = hv[jj];
    }
    wait_async();
    __syncthreads();

    // S = (Q @ K^T) * scale
    v8f sacc[4] = {};
#pragma unroll
    for (int j = 0; j < 4; j++) {
      const _Float16* kr = &Kt[j * 16 + mr][0];
      v16h b0 = frag_from(kr + hi * 8, kr + 16 + hi * 8);
      v16h b1 = frag_from(kr + 32 + hi * 8, kr + 48 + hi * 8);
      sacc[j] = __builtin_amdgcn_wmma_f32_16x16x32_f16(false, aq[0], false, b0,
                                                       (short)0, sacc[j], false, false);
      sacc[j] = __builtin_amdgcn_wmma_f32_16x16x32_f16(false, aq[1], false, b1,
                                                       (short)0, sacc[j], false, false);
    }
    const int rowb = q0 + wr * 16 + hi * 8;
#pragma unroll
    for (int j = 0; j < 4; j++)
#pragma unroll
      for (int r = 0; r < 8; r++) sacc[j][r] *= 0.125f;
    if (kc0 + FA_BK - 1 > rowb) {
#pragma unroll
      for (int j = 0; j < 4; j++) {
        const int col = kc0 + j * 16 + mr;
#pragma unroll
        for (int r = 0; r < 8; r++)
          if (col > rowb + r) sacc[j][r] = -1.0e30f;
      }
    }

    // Online softmax update.
    float pm[8];
#pragma unroll
    for (int r = 0; r < 8; r++)
      pm[r] = fmaxf(fmaxf(sacc[0][r], sacc[1][r]), fmaxf(sacc[2][r], sacc[3][r]));
#pragma unroll
    for (int r = 0; r < 8; r++)
#pragma unroll
      for (int msk = 1; msk < 16; msk <<= 1)
        pm[r] = fmaxf(pm[r], __shfl_xor(pm[r], msk, 32));

    float alpha[8];
#pragma unroll
    for (int r = 0; r < 8; r++) {
      float nm = fmaxf(mrow[r], pm[r]);
      alpha[r] = __expf(mrow[r] - nm);
      mrow[r] = nm;
    }
    float rs[8] = {};
#pragma unroll
    for (int j = 0; j < 4; j++)
#pragma unroll
      for (int r = 0; r < 8; r++) {
        float p = __expf(sacc[j][r] - mrow[r]);
        sacc[j][r] = p;
        rs[r] += p;
      }
#pragma unroll
    for (int r = 0; r < 8; r++) {
#pragma unroll
      for (int msk = 1; msk < 16; msk <<= 1) rs[r] += __shfl_xor(rs[r], msk, 32);
      lrow[r] = lrow[r] * alpha[r] + rs[r];
    }
#pragma unroll
    for (int j2 = 0; j2 < 4; j2++)
#pragma unroll
      for (int r = 0; r < 8; r++) po[j2][r] *= alpha[r];

    // P (C-layout) -> per-wave LDS tile.
#pragma unroll
    for (int j = 0; j < 4; j++) {
      const int col = j * 16 + mr;
#pragma unroll
      for (int r = 0; r < 8; r++)
        QP[wr * 16 + hi * 8 + r][col] = (_Float16)sacc[j][r];
    }
    __syncthreads();

    // O += P @ V
    const _Float16* pr = &QP[wr * 16 + mr][0];
#pragma unroll
    for (int k2 = 0; k2 < 2; k2++) {
      v16h ap = frag_from(pr + k2 * 32 + hi * 8, pr + k2 * 32 + 16 + hi * 8);
#pragma unroll
      for (int j2 = 0; j2 < 4; j2++) {
        const _Float16* vr = &Vt[j2 * 16 + mr][0];
        v16h bv = frag_from(vr + k2 * 32 + hi * 8, vr + k2 * 32 + 16 + hi * 8);
        po[j2] = __builtin_amdgcn_wmma_f32_16x16x32_f16(false, ap, false, bv,
                                                        (short)0, po[j2], false, false);
      }
    }
  }

  // O /= l, write f16 [B,S,H,DK]
  _Float16* Op = O + (size_t)b * cS * cD + h * cDK;
#pragma unroll
  for (int r = 0; r < 8; r++) {
    const float inv = 1.f / lrow[r];
    const int row = q0 + wr * 16 + hi * 8 + r;
#pragma unroll
    for (int j2 = 0; j2 < 4; j2++)
      Op[(size_t)row * cD + j2 * 16 + mr] = (_Float16)(po[j2][r] * inv);
  }
}

// LayerNorm over D per row; writes f32 (residual stream) + f16 (GEMM A copy).
__global__ __launch_bounds__(NT) void layernorm_k(const float* __restrict__ X,
                                                  const float* __restrict__ g,
                                                  const float* __restrict__ bt,
                                                  float* __restrict__ Y,
                                                  _Float16* __restrict__ Y16) {
  __shared__ float r1[NT], r2[NT];
  const int row = blockIdx.x;
  const float* x = X + (size_t)row * cD;
  float* y = Y + (size_t)row * cD;
  _Float16* y16 = Y16 + (size_t)row * cD;
  const int tid = threadIdx.x;
  float s = 0.f, s2 = 0.f;
  for (int d = tid; d < cD; d += NT) {
    float v = x[d];
    s += v;
    s2 += v * v;
  }
  r1[tid] = s;
  r2[tid] = s2;
  __syncthreads();
  for (int st = NT / 2; st > 0; st >>= 1) {
    if (tid < st) {
      r1[tid] += r1[tid + st];
      r2[tid] += r2[tid + st];
    }
    __syncthreads();
  }
  const float mean = r1[0] / (float)cD;
  const float var = r2[0] / (float)cD - mean * mean;
  const float rstd = rsqrtf(var + 1e-5f);
  for (int d = tid; d < cD; d += NT) {
    float v = (x[d] - mean) * rstd * g[d] + bt[d];
    y[d] = v;
    y16[d] = (_Float16)v;
  }
}

// Token embedding + sinusoidal positional encoding; writes f32 + f16.
__global__ __launch_bounds__(NT) void embed_pe_k(const int* __restrict__ ids,
                                                 const float* __restrict__ emb,
                                                 float* __restrict__ X,
                                                 _Float16* __restrict__ X16) {
  const int row = blockIdx.x;  // b*S + s
  const int s = row & (cS - 1);
  const int id = ids[row];
  const float* e = emb + (size_t)id * cD;
  float* xr = X + (size_t)row * cD;
  _Float16* xh = X16 + (size_t)row * cD;
  for (int d = threadIdx.x; d < cD; d += NT) {
    int i = d >> 1;
    float freq = __expf(-logf(10000.f) * (2.f * (float)i) / (float)cD);
    float ang = (float)s * freq;
    float pe = (d & 1) ? __cosf(ang) : __sinf(ang);
    float v = e[d] + pe;
    xr[d] = v;
    xh[d] = (_Float16)v;
  }
}

// ---------------------------------------------------------------------------
extern "C" void kernel_launch(void* const* d_in, const int* in_sizes, int n_in,
                              void* d_out, int out_size, void* d_ws, size_t ws_size,
                              hipStream_t stream) {
  const int* ids = (const int*)d_in[0];
  const float* emb = (const float*)d_in[1];
  const float* Wq = (const float*)d_in[2];
  const float* Wk = (const float*)d_in[3];
  const float* Wv = (const float*)d_in[4];
  const float* Wo = (const float*)d_in[5];
  const float* bq = (const float*)d_in[6];
  const float* bk = (const float*)d_in[7];
  const float* bv = (const float*)d_in[8];
  const float* bo = (const float*)d_in[9];
  const float* ln1g = (const float*)d_in[10];
  const float* ln1b = (const float*)d_in[11];
  const float* W1 = (const float*)d_in[12];
  const float* b1 = (const float*)d_in[13];
  const float* W2 = (const float*)d_in[14];
  const float* b2 = (const float*)d_in[15];
  const float* ln2g = (const float*)d_in[16];
  const float* ln2b = (const float*)d_in[17];
  const float* lnfg = (const float*)d_in[18];
  const float* lnfb = (const float*)d_in[19];
  const float* Wout = (const float*)d_in[20];
  const float* bout = (const float*)d_in[21];
  float* out = (float*)d_out;

  const size_t M = (size_t)cB * cS;  // 2048 token rows
  uint8_t* wp = (uint8_t*)d_ws;
  auto carve = [&](size_t bytes) -> void* {
    void* p = (void*)wp;
    wp += (bytes + 255) & ~(size_t)255;
    return p;
  };
  float* x = (float*)carve(M * cD * 4);  // residual stream (f32)
  float* t = (float*)carve(M * cD * 4);  // pre-LN temp (f32)
  _Float16* x16 = (_Float16*)carve(M * cD * 2);
  _Float16* t16 = (_Float16*)carve(M * cD * 2);
  _Float16* q16 = (_Float16*)carve(M * cD * 2);
  _Float16* k16 = (_Float16*)carve(M * cD * 2);
  _Float16* v16 = (_Float16*)carve(M * cD * 2);
  _Float16* o16 = (_Float16*)carve(M * cD * 2);
  _Float16* h16 = (_Float16*)carve(M * cFF * 2);

  const dim3 blk(NT);
  const dim3 gDD((unsigned)(M / BM), cD / BN);  // 16 x 8

  embed_pe_k<<<dim3((unsigned)M), blk, 0, stream>>>(ids, emb, x, x16);

  for (int l = 0; l < cL; l++) {
    const float* Wq_l = Wq + (size_t)l * cD * cD;
    const float* Wk_l = Wk + (size_t)l * cD * cD;
    const float* Wv_l = Wv + (size_t)l * cD * cD;
    const float* Wo_l = Wo + (size_t)l * cD * cD;
    const float* W1_l = W1 + (size_t)l * cD * cFF;
    const float* W2_l = W2 + (size_t)l * cFF * cD;
    const float* bq_l = bq + (size_t)l * cD;
    const float* bk_l = bk + (size_t)l * cD;
    const float* bv_l = bv + (size_t)l * cD;
    const float* bo_l = bo + (size_t)l * cD;
    const float* b1_l = b1 + (size_t)l * cFF;
    const float* b2_l = b2 + (size_t)l * cD;

    // QKV projections (f16 outputs feed flash directly)
    gemm_wmma_k<0, true><<<gDD, blk, 0, stream>>>(x16, cD, Wq_l, cD, q16, cD, bq_l,
                                                  nullptr, cD);
    gemm_wmma_k<0, true><<<gDD, blk, 0, stream>>>(x16, cD, Wk_l, cD, k16, cD, bk_l,
                                                  nullptr, cD);
    gemm_wmma_k<0, true><<<gDD, blk, 0, stream>>>(x16, cD, Wv_l, cD, v16, cD, bv_l,
                                                  nullptr, cD);

    // Fused causal attention
    flash_attn_k<<<dim3(cS / FA_BQ, cB * cH), blk, 0, stream>>>(q16, k16, v16, o16);

    // Output projection + residual (f32), then LN1
    gemm_wmma_k<2, false><<<gDD, blk, 0, stream>>>(o16, cD, Wo_l, cD, t, cD, bo_l, x,
                                                   cD);
    layernorm_k<<<dim3((unsigned)M), blk, 0, stream>>>(t, ln1g + (size_t)l * cD,
                                                       ln1b + (size_t)l * cD, x, x16);

    // FFN: ReLU(x@W1+b1) (f16) @W2+b2 + residual, then LN2
    gemm_wmma_k<1, true><<<dim3((unsigned)(M / BM), cFF / BN), blk, 0, stream>>>(
        x16, cD, W1_l, cFF, h16, cFF, b1_l, nullptr, cD);
    gemm_wmma_k<2, false><<<gDD, blk, 0, stream>>>(h16, cFF, W2_l, cD, t, cD, b2_l, x,
                                                   cFF);
    layernorm_k<<<dim3((unsigned)M), blk, 0, stream>>>(t, ln2g + (size_t)l * cD,
                                                       ln2b + (size_t)l * cD, x, x16);
  }

  // Final LN + logits
  layernorm_k<<<dim3((unsigned)M), blk, 0, stream>>>(x, lnfg, lnfb, t, t16);
  gemm_wmma_k<0, false><<<dim3((unsigned)(M / BM), cV / BN), blk, 0, stream>>>(
      t16, cD, Wout, cV, out, cV, bout, nullptr, cD);
}
